// ComputeLoss_50732153700580
// MI455X (gfx1250) — matrix operand order repaired and checked by reading, not compiled, not run on hardware
//
#include <hip/hip_runtime.h>
#include <math.h>

#define NB   16      // batch
#define NA   8400    // anchors (6400 + 1600 + 400)
#define NCH  144     // channels per anchor (64 dist + 80 cls)
#define NCC  80      // classes
#define NMB  32      // max boxes per image
#define NTOP 10

typedef __attribute__((ext_vector_type(16))) _Float16 v16h;
typedef __attribute__((ext_vector_type(8)))  float    v8f;

// ---------- helpers ----------
__device__ __forceinline__ const float* chan_base(const float* p3, const float* p4,
                                                  const float* p5, int b, int a) {
  if (a < 6400) return p3 + ((size_t)b * 6400 + a) * NCH;
  if (a < 8000) return p4 + ((size_t)b * 1600 + (a - 6400)) * NCH;
  return p5 + ((size_t)b * 400 + (a - 8000)) * NCH;
}

__device__ __forceinline__ void anchor_pos(int a, float& s, float& ax, float& ay) {
  int local, w;
  if (a < 6400)      { s = 8.f;  local = a;        w = 80; }
  else if (a < 8000) { s = 16.f; local = a - 6400; w = 40; }
  else               { s = 32.f; local = a - 8000; w = 20; }
  ax = (float)(local % w) + 0.5f;
  ay = (float)(local / w) + 0.5f;
}

__device__ __forceinline__ float ciou_f(float b1x1, float b1y1, float b1x2, float b1y2,
                                        float b2x1, float b2y1, float b2x2, float b2y2) {
  const float eps = 1e-7f;
  float w1 = b1x2 - b1x1, h1 = b1y2 - b1y1 + eps;
  float w2 = b2x2 - b2x1, h2 = b2y2 - b2y1 + eps;
  float iw = fminf(b1x2, b2x2) - fmaxf(b1x1, b2x1);
  float ih = fminf(b1y2, b2y2) - fmaxf(b1y1, b2y1);
  float inter = fmaxf(iw, 0.f) * fmaxf(ih, 0.f);
  float uni = w1 * h1 + w2 * h2 - inter + eps;
  float iou = inter / uni;
  float cw  = fmaxf(b1x2, b2x2) - fminf(b1x1, b2x1);
  float chh = fmaxf(b1y2, b2y2) - fminf(b1y1, b2y1);
  float c2  = cw * cw + chh * chh + eps;
  float dx = b2x1 + b2x2 - b1x1 - b1x2;
  float dy = b2y1 + b2y2 - b1y1 - b1y2;
  float rho2 = (dx * dx + dy * dy) * 0.25f;
  float dat = atanf(w2 / h2) - atanf(w1 / h1);
  const float inv_pi2 = 4.0f / 9.869604401089358f;  // 4/pi^2
  float v = inv_pi2 * dat * dat;
  float al = v / (v - iou + 1.0f + eps);
  return iou - (rho2 / c2 + v * al);
}

// ---------- K0: build GT tensors ----------
__global__ void k_gt(const float* __restrict__ tg, int nt,
                     float* __restrict__ gtraw, float* __restrict__ gtbox,
                     int* __restrict__ gtlab, float* __restrict__ gtmask) {
  int tid = threadIdx.x;
  for (int i = tid; i < NB * NMB * 5; i += 256) gtraw[i] = 0.f;
  __syncthreads();
  if (tid < nt) {
    int img = (int)tg[tid * 6];
    int rank = 0;
    for (int t2 = 0; t2 < tid; t2++)
      if ((int)tg[t2 * 6] == img) rank++;
    if (img >= 0 && img < NB && rank < NMB)
      for (int j = 0; j < 5; j++)
        gtraw[(img * NMB + rank) * 5 + j] = tg[tid * 6 + 1 + j];
  }
  __syncthreads();
  for (int p = tid; p < NB * NMB; p += 256) {
    float cls = gtraw[p * 5 + 0];
    float cx = gtraw[p * 5 + 1] * 640.f, cy = gtraw[p * 5 + 2] * 640.f;
    float w  = gtraw[p * 5 + 3] * 640.f, h  = gtraw[p * 5 + 4] * 640.f;
    float x1 = cx - w * 0.5f, y1 = cy - h * 0.5f;
    float x2 = cx + w * 0.5f, y2 = cy + h * 0.5f;
    gtbox[p * 4 + 0] = x1; gtbox[p * 4 + 1] = y1;
    gtbox[p * 4 + 2] = x2; gtbox[p * 4 + 3] = y2;
    gtlab[p]  = (int)cls;
    gtmask[p] = ((x1 + y1 + x2 + y2) > 0.f) ? 1.f : 0.f;
  }
}

// ---------- K1: DFL decode via WMMA ----------
// Each wave decodes 16 anchors. A (16x32 f16): rows = anchors, K = [side0 bins | side1 bins].
// B (32x16 f16): column 0 = project vector for K<16, column 1 = project for K>=16.
// D (16x16 f32): col0 = expectation side s0, col1 = side s1.
__global__ void k_decode(const float* __restrict__ p3, const float* __restrict__ p4,
                         const float* __restrict__ p5, float* __restrict__ predb) {
  int tid = threadIdx.x, w = tid >> 5, lane = tid & 31;
  int m = lane & 15, half = lane >> 4;
  int g0 = (blockIdx.x * 8 + w) * 16;  // 16-anchor group; aligned to b/level boundaries
  int b  = g0 / NA;
  int a0 = g0 % NA;
  int a  = a0 + m;
  const float* cb = chan_base(p3, p4, p5, b, a);

  __shared__ float pdres[8][16][4];

  // B fragment: table order -> element j<8 holds K=8*half+j, element j>=8 holds K=16+8*half+(j-8).
  int bn = lane & 15;
  v16h bfrag;
#pragma unroll
  for (int j = 0; j < 16; j++) {
    float v = 0.f;
    if (bn == 0 && j < 8) v = (float)(half * 8 + j);
    if (bn == 1 && j >= 8) v = (float)(half * 8 + j - 8);
    bfrag[j] = (_Float16)v;
  }

#pragma unroll
  for (int sp = 0; sp < 2; sp++) {
    int s0 = sp * 2, s1 = sp * 2 + 1;
    float x0[8], x1[8];
#pragma unroll
    for (int j = 0; j < 8; j++) {
      x0[j] = cb[s0 * 16 + half * 8 + j];
      x1[j] = cb[s1 * 16 + half * 8 + j];
    }
    // softmax over 16 bins split across lane pair (L, L^16)
    float m0 = x0[0], m1 = x1[0];
#pragma unroll
    for (int j = 1; j < 8; j++) { m0 = fmaxf(m0, x0[j]); m1 = fmaxf(m1, x1[j]); }
    m0 = fmaxf(m0, __shfl_xor(m0, 16));
    m1 = fmaxf(m1, __shfl_xor(m1, 16));
    float sum0 = 0.f, sum1 = 0.f;
#pragma unroll
    for (int j = 0; j < 8; j++) {
      x0[j] = expf(x0[j] - m0); sum0 += x0[j];
      x1[j] = expf(x1[j] - m1); sum1 += x1[j];
    }
    sum0 += __shfl_xor(sum0, 16);
    sum1 += __shfl_xor(sum1, 16);
    float r0 = 1.f / sum0, r1 = 1.f / sum1;

    v16h afrag;
#pragma unroll
    for (int j = 0; j < 8; j++) {
      afrag[j]     = (_Float16)(x0[j] * r0);
      afrag[8 + j] = (_Float16)(x1[j] * r1);
    }
    v8f c = {};
    c = __builtin_amdgcn_wmma_f32_16x16x32_f16(false, afrag, false, bfrag,
                                               (short)0, c, false, false);
    int N = lane & 15;
    if (N < 2) {
#pragma unroll
      for (int r = 0; r < 8; r++)
        pdres[w][r + half * 8][sp * 2 + N] = c[r];
    }
    __syncthreads();
  }

  if (lane < 16) {
    int aa = a0 + lane;
    float s, ax, ay;
    anchor_pos(aa, s, ax, ay);
    float4 out;
    out.x = ax - pdres[w][lane][0];
    out.y = ay - pdres[w][lane][1];
    out.z = ax + pdres[w][lane][2];
    out.w = ay + pdres[w][lane][3];
    ((float4*)predb)[(size_t)b * NA + aa] = out;
  }
}

// ---------- K2: overlaps + align + top-k per (b, gt) ----------
__global__ void k_metrics(const float* __restrict__ p3, const float* __restrict__ p4,
                          const float* __restrict__ p5, const float* __restrict__ predb,
                          const float* __restrict__ gtbox, const int* __restrict__ gtlab,
                          const float* __restrict__ gtmask,
                          float* __restrict__ ovl, float* __restrict__ alg,
                          int* __restrict__ topidx, int* __restrict__ topval) {
  int bn = blockIdx.x, bi = bn >> 5;
  int tid = threadIdx.x;
  float gx1 = gtbox[bn * 4 + 0], gy1 = gtbox[bn * 4 + 1];
  float gx2 = gtbox[bn * 4 + 2], gy2 = gtbox[bn * 4 + 3];
  float gmask = gtmask[bn];
  int glab = gtlab[bn];
  const float4* pb4 = (const float4*)predb;

  for (int a = tid; a < NA; a += 256) {
    __builtin_prefetch(&pb4[(size_t)bi * NA + a + 256], 0, 3);
    float s, ax, ay;
    anchor_pos(a, s, ax, ay);
    float4 pb = pb4[(size_t)bi * NA + a];
    float ci = ciou_f(gx1, gy1, gx2, gy2, pb.x * s, pb.y * s, pb.z * s, pb.w * s);
    float o = fmaxf(ci, 0.f);
    ovl[(size_t)bn * NA + a] = o;
    float logit = chan_base(p3, p4, p5, bi, a)[64 + glab];
    float sc = 1.f / (1.f + expf(-logit));
    float o3 = o * o * o;
    alg[(size_t)bn * NA + a] = sqrtf(sc) * o3 * o3;  // score^0.5 * ovl^6
  }
  __syncthreads();

  __shared__ float sv[256];
  __shared__ int   si[256];
  __shared__ int   picked[NTOP];
  __shared__ float top0;
  for (int k = 0; k < NTOP; k++) {
    float bv = -1.f;
    int bidx = 0x7fffffff;
    for (int a = tid; a < NA; a += 256) {
      bool ex = false;
      for (int j = 0; j < k; j++)
        if (picked[j] == a) { ex = true; break; }
      if (ex) continue;
      float s, ax, ay;
      anchor_pos(a, s, ax, ay);
      float axp = ax * s, ayp = ay * s;
      float mn = fminf(fminf(axp - gx1, ayp - gy1), fminf(gx2 - axp, gy2 - ayp));
      float mv = (gmask > 0.f && mn > 1e-9f) ? alg[(size_t)bn * NA + a] : 0.f;
      if (mv > bv || (mv == bv && a < bidx)) { bv = mv; bidx = a; }
    }
    sv[tid] = bv; si[tid] = bidx;
    __syncthreads();
    for (int off = 128; off > 0; off >>= 1) {
      if (tid < off) {
        float v2 = sv[tid + off]; int i2 = si[tid + off];
        if (v2 > sv[tid] || (v2 == sv[tid] && i2 < si[tid])) { sv[tid] = v2; si[tid] = i2; }
      }
      __syncthreads();
    }
    if (tid == 0) { picked[k] = si[0]; if (k == 0) top0 = sv[0]; }
    __syncthreads();
  }
  if (tid < NTOP) topidx[bn * NTOP + tid] = picked[tid];
  if (tid == 0)   topval[bn] = (top0 > 1e-9f) ? 1 : 0;
}

// ---------- K3a: per-anchor assignment resolution ----------
__global__ void k_resolve(const float* __restrict__ gtbox, const float* __restrict__ gtmask,
                          const int* __restrict__ topidx, const int* __restrict__ topval,
                          const float* __restrict__ ovl, int* __restrict__ chosen) {
  int i = blockIdx.x * 256 + threadIdx.x;
  int b = i / NA, a = i % NA;
  float s, ax, ay;
  anchor_pos(a, s, ax, ay);
  float axp = ax * s, ayp = ay * s;
  int cnt = 0, first = -1;
  for (int n = 0; n < NMB; n++) {
    int bn = b * NMB + n;
    if (!topval[bn]) continue;
    bool in = false;
    for (int k = 0; k < NTOP; k++)
      if (topidx[bn * NTOP + k] == a) { in = true; break; }
    if (!in) continue;
    if (gtmask[bn] <= 0.f) continue;
    float mn = fminf(fminf(axp - gtbox[bn * 4 + 0], ayp - gtbox[bn * 4 + 1]),
                     fminf(gtbox[bn * 4 + 2] - axp, gtbox[bn * 4 + 3] - ayp));
    if (!(mn > 1e-9f)) continue;
    cnt++;
    if (first < 0) first = n;
  }
  int ch = -1;
  if (cnt == 1) ch = first;
  else if (cnt > 1) {  // multi: pick gt with max overlap (first max)
    float bo = -1e30f; int bnn = 0;
    for (int n = 0; n < NMB; n++) {
      float o = ovl[(size_t)(b * NMB + n) * NA + a];
      if (o > bo) { bo = o; bnn = n; }
    }
    ch = bnn;
  }
  chosen[i] = ch;
}

// ---------- K3b: per-gt max of masked align / overlap ----------
__global__ void k_posmax(const float* __restrict__ alg, const float* __restrict__ ovl,
                         const int* __restrict__ chosen,
                         float* __restrict__ posA, float* __restrict__ posO) {
  int bn = blockIdx.x, bi = bn >> 5, ni = bn & 31;
  int tid = threadIdx.x;
  float ma = 0.f, mo = 0.f;
  for (int a = tid; a < NA; a += 256) {
    if (chosen[bi * NA + a] == ni) {
      ma = fmaxf(ma, alg[(size_t)bn * NA + a]);
      mo = fmaxf(mo, ovl[(size_t)bn * NA + a]);
    }
  }
  __shared__ float sa[256], so[256];
  sa[tid] = ma; so[tid] = mo;
  __syncthreads();
  for (int off = 128; off > 0; off >>= 1) {
    if (tid < off) { sa[tid] = fmaxf(sa[tid], sa[tid + off]); so[tid] = fmaxf(so[tid], so[tid + off]); }
    __syncthreads();
  }
  if (tid == 0) { posA[bn] = sa[0]; posO[bn] = so[0]; }
}

// ---------- K3c: per-anchor loss terms, block partial sums ----------
__global__ void k_loss(const float* __restrict__ p3, const float* __restrict__ p4,
                       const float* __restrict__ p5, const float* __restrict__ predb,
                       const float* __restrict__ gtbox, const int* __restrict__ gtlab,
                       const int* __restrict__ chosen, const float* __restrict__ alg,
                       const float* __restrict__ posA, const float* __restrict__ posO,
                       float* __restrict__ partials) {
  int tid = threadIdx.x;
  int i = blockIdx.x * 256 + tid;
  int b = i / NA, a = i % NA;
  const float* cb = chan_base(p3, p4, p5, b, a);
  int ch = chosen[i];
  bool fg = ch >= 0;
  float norm = 0.f;
  int tl = 0;
  if (fg) {
    int bn = b * NMB + ch;
    norm = alg[(size_t)bn * NA + a] * posO[bn] / (posA[bn] + 1e-9f);
    tl = gtlab[bn];
  }
  float lbce = 0.f;
  for (int c = 0; c < NCC; c++) {
    float l = cb[64 + c];
    float z = (fg && c == tl) ? norm : 0.f;
    lbce += fmaxf(l, 0.f) - l * z + log1pf(expf(-fabsf(l)));
  }
  float lbox = 0.f, ldfl = 0.f;
  if (fg) {
    int bn = b * NMB + ch;
    float s, ax, ay;
    anchor_pos(a, s, ax, ay);
    float inv = 1.f / s;
    float tx1 = gtbox[bn * 4 + 0] * inv, ty1 = gtbox[bn * 4 + 1] * inv;
    float tx2 = gtbox[bn * 4 + 2] * inv, ty2 = gtbox[bn * 4 + 3] * inv;
    float4 pb = ((const float4*)predb)[(size_t)b * NA + a];
    float iou = ciou_f(pb.x, pb.y, pb.z, pb.w, tx1, ty1, tx2, ty2);
    lbox = (1.f - iou) * norm;
    float ltrb[4] = {ax - tx1, ay - ty1, tx2 - ax, ty2 - ay};
    float acc = 0.f;
    for (int sd = 0; sd < 4; sd++) {
      float val = fminf(fmaxf(ltrb[sd], 0.f), 14.99f);
      float mx = cb[sd * 16];
      for (int j = 1; j < 16; j++) mx = fmaxf(mx, cb[sd * 16 + j]);
      float sum = 0.f;
      for (int j = 0; j < 16; j++) sum += expf(cb[sd * 16 + j] - mx);
      float lse = mx + logf(sum);
      int t = (int)val;
      float wl = (float)(t + 1) - val;
      float lpl = cb[sd * 16 + t] - lse;
      float lpr = cb[sd * 16 + t + 1] - lse;
      acc += (-lpl) * wl + (-lpr) * (1.f - wl);
    }
    ldfl = acc * 0.25f * norm;
  }
  float lnorm = fg ? norm : 0.f;

  __shared__ float red[256];
  float vals[4] = {lbce, lbox, ldfl, lnorm};
  for (int q = 0; q < 4; q++) {
    red[tid] = vals[q];
    __syncthreads();
    for (int off = 128; off > 0; off >>= 1) {
      if (tid < off) red[tid] += red[tid + off];
      __syncthreads();
    }
    if (tid == 0) partials[blockIdx.x * 4 + q] = red[0];
    __syncthreads();
  }
}

// ---------- K4: deterministic final reduction ----------
__global__ void k_final(const float* __restrict__ partials, int nblocks,
                        float* __restrict__ out) {
  int tid = threadIdx.x;
  float acc[4] = {0.f, 0.f, 0.f, 0.f};
  for (int j = tid; j < nblocks; j += 256)
    for (int q = 0; q < 4; q++) acc[q] += partials[j * 4 + q];
  __shared__ float red[256];
  float tot[4];
  for (int q = 0; q < 4; q++) {
    red[tid] = acc[q];
    __syncthreads();
    for (int off = 128; off > 0; off >>= 1) {
      if (tid < off) red[tid] += red[tid + off];
      __syncthreads();
    }
    tot[q] = red[0];
    __syncthreads();
  }
  if (tid == 0) {
    float tss = fmaxf(tot[3], 1.0f);
    out[0] = (0.5f * tot[0] + 7.5f * tot[1] + 1.5f * tot[2]) / tss;
  }
}

extern "C" void kernel_launch(void* const* d_in, const int* in_sizes, int n_in,
                              void* d_out, int out_size, void* d_ws, size_t ws_size,
                              hipStream_t stream) {
  (void)n_in; (void)out_size; (void)ws_size;
  const float* p3 = (const float*)d_in[0];
  const float* p4 = (const float*)d_in[1];
  const float* p5 = (const float*)d_in[2];
  const float* tg = (const float*)d_in[3];
  int nt = in_sizes[3] / 6;

  char* w = (char*)d_ws;
  size_t off = 0;
  auto alloc = [&](size_t bytes) -> void* {
    void* p = w + off;
    off += (bytes + 255) & ~(size_t)255;
    return p;
  };
  float* gtraw  = (float*)alloc((size_t)NB * NMB * 5 * 4);
  float* gtbox  = (float*)alloc((size_t)NB * NMB * 4 * 4);
  int*   gtlab  = (int*)  alloc((size_t)NB * NMB * 4);
  float* gtmask = (float*)alloc((size_t)NB * NMB * 4);
  float* predb  = (float*)alloc((size_t)NB * NA * 4 * 4);
  float* ovl    = (float*)alloc((size_t)NB * NMB * NA * 4);
  float* alg    = (float*)alloc((size_t)NB * NMB * NA * 4);
  int*   topidx = (int*)  alloc((size_t)NB * NMB * NTOP * 4);
  int*   topval = (int*)  alloc((size_t)NB * NMB * 4);
  int*   chosen = (int*)  alloc((size_t)NB * NA * 4);
  float* posA   = (float*)alloc((size_t)NB * NMB * 4);
  float* posO   = (float*)alloc((size_t)NB * NMB * 4);
  const int NLB = (NB * NA) / 256;  // 525
  float* partials = (float*)alloc((size_t)NLB * 4 * 4);

  k_gt<<<1, 256, 0, stream>>>(tg, nt, gtraw, gtbox, gtlab, gtmask);
  k_decode<<<(NB * NA) / 128, 256, 0, stream>>>(p3, p4, p5, predb);
  k_metrics<<<NB * NMB, 256, 0, stream>>>(p3, p4, p5, predb, gtbox, gtlab, gtmask,
                                          ovl, alg, topidx, topval);
  k_resolve<<<NLB, 256, 0, stream>>>(gtbox, gtmask, topidx, topval, ovl, chosen);
  k_posmax<<<NB * NMB, 256, 0, stream>>>(alg, ovl, chosen, posA, posO);
  k_loss<<<NLB, 256, 0, stream>>>(p3, p4, p5, predb, gtbox, gtlab, chosen, alg,
                                  posA, posO, partials);
  k_final<<<1, 256, 0, stream>>>(partials, NLB, (float*)d_out);
}
